// SuperPointDecoder_11364483465758
// MI455X (gfx1250) — compile-verified, matrix operand order checked by reading, not run on hardware
//
#include <hip/hip_runtime.h>
#include <hip/hip_bf16.h>

typedef __attribute__((ext_vector_type(16))) __bf16 v16bf;
typedef __attribute__((ext_vector_type(8)))  float  v8f;

union BF16x16 { v16bf v; __bf16 e[16]; };
union F32x8   { v8f v; float e[8]; };

constexpr int Bb  = 4;
constexpr int Np  = 25000;
constexpr int Dm  = 256;
constexpr int Hh  = 8;
constexpr int Ll  = 6;
constexpr int NQn = 100;
constexpr int NCo = 19;           // NC+1
constexpr int HID = 1024;
constexpr int BN  = Bb * Np;      // 100000
constexpr int CHUNK = 128;
constexpr int NCH = (Np + CHUNK - 1) / CHUNK;   // 196
constexpr float SCALE = 0.17677669529663687f;   // 1/sqrt(32)

__device__ __forceinline__ v8f wmma_bf16(v16bf a, v16bf b, v8f c) {
  return __builtin_amdgcn_wmma_f32_16x16x32_bf16(false, a, false, b, (short)0, c, false, false);
}

// A fragment (16x32 bf16) from LDS row-major buffer
__device__ __forceinline__ v16bf afrag_lds(const __bf16* base, int row0, int stride, int k0) {
  int lane = threadIdx.x & 31;
  int row = row0 + (lane & 15);
  int k = k0 + ((lane >> 4) << 4);
  return *(const v16bf*)(base + row * stride + k);
}

// A fragment from f32 LDS (convert to bf16)
__device__ __forceinline__ v16bf afrag_ldsf(const float* base, int row0, int stride, int k0) {
  int lane = threadIdx.x & 31;
  const float* p = base + (row0 + (lane & 15)) * stride + k0 + ((lane >> 4) << 4);
  BF16x16 r;
#pragma unroll
  for (int i = 0; i < 16; ++i) r.e[i] = (__bf16)p[i];
  return r.v;
}

// B fragment (32x16): B[k][n] = mat[n][k], mat row-major bf16, with row bound
__device__ __forceinline__ v16bf bfrag_rowk(const __bf16* mat, int n0, int rowstride, int k0, int nmax) {
  int lane = threadIdx.x & 31;
  int n = n0 + (lane & 15);
  int k = k0 + ((lane >> 4) << 4);
  if (n < nmax) return *(const v16bf*)(mat + (size_t)n * rowstride + k);
  BF16x16 z;
#pragma unroll
  for (int i = 0; i < 16; ++i) z.e[i] = (__bf16)0.0f;
  return z.v;
}

// B fragment from f32 weight matrix W[n][k] (convert to bf16)
__device__ __forceinline__ v16bf bfrag_w32(const float* w, int n0, int Kst, int k0) {
  int lane = threadIdx.x & 31;
  int n = n0 + (lane & 15);
  const float* p = w + (size_t)n * Kst + k0 + ((lane >> 4) << 4);
  BF16x16 r;
#pragma unroll
  for (int i = 0; i < 16; ++i) r.e[i] = (__bf16)p[i];
  return r.v;
}

// B fragment for V: B[k][n] = vb[(jbase+k)*Dm + hoff + n]
__device__ __forceinline__ v16bf bfrag_vglobal(const __bf16* vb, int jbase, int k0, int hoff, int n0, int jmax) {
  int lane = threadIdx.x & 31;
  int n = n0 + (lane & 15);
  int ks = k0 + ((lane >> 4) << 4);
  BF16x16 r;
#pragma unroll
  for (int i = 0; i < 16; ++i) {
    int j = jbase + ks + i;
    r.e[i] = (j < jmax) ? vb[(size_t)j * Dm + hoff + n] : (__bf16)0.0f;
  }
  return r.v;
}

// ---------------- Kernel 1: input_proj (Linear+LN+ReLU) + x_mask MLP -> bf16 ----------------
__global__ void k_inproj(const float* __restrict__ x,
                         const float* __restrict__ ipw, const float* __restrict__ ipb,
                         const float* __restrict__ lng, const float* __restrict__ lnb,
                         const float* __restrict__ w1, const float* __restrict__ b1,
                         const float* __restrict__ w2, const float* __restrict__ b2,
                         __bf16* __restrict__ inst, __bf16* __restrict__ maskf) {
  __shared__ float xs[16 * 32];
  __shared__ float fs[16 * 256];
  __shared__ float hs[16 * 256];
  __shared__ float mrow[16], rrow[16];
  int t = threadIdx.x;
  int rowbase = blockIdx.x * 16;
  for (int i = t; i < 16 * 32; i += 256) {
    int r = i >> 5, c = i & 31;
    xs[i] = x[(size_t)(rowbase + r) * 32 + c];
  }
  __syncthreads();
  int col = t;
  {
    float a0[16], a1[16];
#pragma unroll
    for (int r = 0; r < 16; ++r) { a0[r] = 0.f; a1[r] = 0.f; }
    for (int k = 0; k < 32; ++k) {
      float wa = ipw[col * 32 + k];
      float wb = w1[col * 32 + k];
#pragma unroll
      for (int r = 0; r < 16; ++r) { float xv = xs[r * 32 + k]; a0[r] += xv * wa; a1[r] += xv * wb; }
    }
    float ba = ipb[col], bb = b1[col];
#pragma unroll
    for (int r = 0; r < 16; ++r) {
      fs[r * 256 + col] = a0[r] + ba;
      float h = a1[r] + bb;
      hs[r * 256 + col] = h > 0.f ? h : 0.f;
    }
  }
  __syncthreads();
  if (t < 16) {
    float m = 0.f, v = 0.f;
    for (int c = 0; c < 256; ++c) { float u = fs[t * 256 + c]; m += u; v += u * u; }
    m *= (1.f / 256.f); v = v * (1.f / 256.f) - m * m;
    mrow[t] = m; rrow[t] = rsqrtf(v + 1e-5f);
  }
  __syncthreads();
  {
    float g = lng[col], bo = lnb[col];
#pragma unroll
    for (int r = 0; r < 16; ++r) {
      float u = (fs[r * 256 + col] - mrow[r]) * rrow[r] * g + bo;
      u = u > 0.f ? u : 0.f;
      inst[(size_t)(rowbase + r) * 256 + col] = (__bf16)u;
    }
  }
  {
    float a[16];
#pragma unroll
    for (int r = 0; r < 16; ++r) a[r] = 0.f;
    for (int k = 0; k < 256; ++k) {
      float wv = w2[col * 256 + k];
#pragma unroll
      for (int r = 0; r < 16; ++r) a[r] += hs[r * 256 + k] * wv;
    }
    float bb = b2[col];
#pragma unroll
    for (int r = 0; r < 16; ++r)
      maskf[(size_t)(rowbase + r) * 256 + col] = (__bf16)(a[r] + bb);
  }
}

// ---------------- Kernel 2: K/V projection (WMMA bf16) ----------------
__global__ void k_kvproj(const __bf16* __restrict__ inst,
                         const float* __restrict__ wk, const float* __restrict__ wv,
                         const float* __restrict__ bk, const float* __restrict__ bv,
                         __bf16* __restrict__ Kb, __bf16* __restrict__ Vb) {
  __shared__ __bf16 As[64 * 256];
  int t = threadIdx.x;
  int rowbase = blockIdx.x * 64;
  {
    const unsigned int* src = (const unsigned int*)inst;
    unsigned int* dst = (unsigned int*)As;
    for (int i = t; i < 64 * 128; i += 256) {
      int r = i >> 7;
      unsigned int val = 0u;
      if (rowbase + r < BN) val = src[(size_t)(rowbase + r) * 128 + (i & 127)];
      dst[i] = val;
    }
  }
  __syncthreads();
  int w = t >> 5, lane = t & 31;
  int rs = (w & 3) * 16;
  int cg = (w >> 2);
  v16bf af[8];
#pragma unroll
  for (int kc = 0; kc < 8; ++kc) af[kc] = afrag_lds(As, rs, 256, kc * 32);
  for (int i = 0; i < 16; ++i) {
    int ct = cg * 16 + i;
    int gc0 = ct * 16;
    const float* wsrc = (gc0 < 256) ? wk : wv;
    const float* bsrc = (gc0 < 256) ? bk : bv;
    __bf16* dst = (gc0 < 256) ? Kb : Vb;
    int cc0 = gc0 & 255;
    F32x8 acc;
#pragma unroll
    for (int j = 0; j < 8; ++j) acc.e[j] = 0.f;
#pragma unroll
    for (int kc = 0; kc < 8; ++kc) {
      v16bf bf = bfrag_w32(wsrc, cc0, 256, kc * 32);
      acc.v = wmma_bf16(af[kc], bf, acc.v);
    }
    int cc = cc0 + (lane & 15);
    float bias = bsrc[cc];
    int row0 = rowbase + rs + ((lane >> 4) << 3);
#pragma unroll
    for (int r = 0; r < 8; ++r) {
      int row = row0 + r;
      if (row < BN) dst[(size_t)row * 256 + cc] = (__bf16)(acc.e[r] + bias);
    }
  }
}

// ---------------- Generic small linear: one block per row ----------------
// out[row, 0..N) = LN?( relu?( X[row]@W.T + bias + res[row] ) )   (W==null -> identity, N==K)
__global__ void k_linear(const float* __restrict__ X, int Kd,
                         const float* __restrict__ W, const float* __restrict__ bias,
                         const float* __restrict__ res,
                         const float* __restrict__ lng, const float* __restrict__ lnb,
                         int N, int dorelu, float* __restrict__ out) {
  __shared__ float xr[1024];
  __shared__ float orow[1024];
  __shared__ float red[2];
  int t = threadIdx.x;
  size_t row = blockIdx.x;
  for (int k = t; k < Kd; k += 256) xr[k] = X[row * Kd + k];
  __syncthreads();
  for (int c = t; c < N; c += 256) {
    float acc = bias ? bias[c] : 0.f;
    if (W) {
      const float* wr = W + (size_t)c * Kd;
      for (int k = 0; k < Kd; ++k) acc += xr[k] * wr[k];
    } else {
      acc += xr[c];
    }
    if (res) acc += res[row * N + c];
    if (dorelu && acc < 0.f) acc = 0.f;
    orow[c] = acc;
  }
  __syncthreads();
  if (lng) {
    if (t == 0) {
      float m = 0.f, v = 0.f;
      for (int c = 0; c < N; ++c) { float u = orow[c]; m += u; v += u * u; }
      m /= (float)N; v = v / (float)N - m * m;
      red[0] = m; red[1] = rsqrtf(v + 1e-5f);
    }
    __syncthreads();
    float m = red[0], rsd = red[1];
    for (int c = t; c < N; c += 256)
      out[row * N + c] = (orow[c] - m) * rsd * lng[c] + lnb[c];
  } else {
    for (int c = t; c < N; c += 256) out[row * N + c] = orow[c];
  }
}

// ---------------- Cross-attention split-K partial (flash style, WMMA) ----------------
__global__ void k_ca_partial(const float* __restrict__ Qlin,
                             const __bf16* __restrict__ Kb, const __bf16* __restrict__ Vb,
                             float* __restrict__ part, float* __restrict__ pm, float* __restrict__ pl) {
  __shared__ float ss[112 * 128];
  __shared__ __bf16 qs[112 * 32];
  __shared__ float ms[112], ls[112];
  int t = threadIdx.x;
  int cidx = blockIdx.x % NCH;
  int bh = blockIdx.x / NCH;
  int b = bh >> 3, h = bh & 7;
  int hoff = h * 32;
  int jbase = cidx * CHUNK;
  for (int i = t; i < 112 * 32; i += 256) {
    int q = i >> 5, d = i & 31;
    float v = 0.f;
    if (q < NQn) v = Qlin[((size_t)(b * NQn + q)) * Dm + hoff + d] * SCALE;
    qs[i] = (__bf16)v;
  }
  __syncthreads();
  int w = t >> 5, lane = t & 31;
  // scores = Q @ K^T : 7 row tiles x 8 col tiles, K=32 (one WMMA each)
  for (int tix = w; tix < 56; tix += 8) {
    int rt = tix >> 3, ct = tix & 7;
    v16bf a = afrag_lds(qs, rt * 16, 32, 0);
    v16bf bm = bfrag_rowk(Kb + (size_t)jbase * Dm + hoff, ct * 16, Dm, 0, Np - jbase);
    F32x8 acc;
#pragma unroll
    for (int j = 0; j < 8; ++j) acc.e[j] = 0.f;
    acc.v = wmma_bf16(a, bm, acc.v);
    int colc = ct * 16 + (lane & 15);
    bool valid = (jbase + colc) < Np;
    int row0 = rt * 16 + ((lane >> 4) << 3);
#pragma unroll
    for (int r = 0; r < 8; ++r)
      ss[(row0 + r) * 128 + colc] = valid ? acc.e[r] : -1e30f;
  }
  __syncthreads();
  if (t < 112) {
    if (t < NQn) {
      float m = -1e30f;
      for (int j = 0; j < 128; ++j) m = fmaxf(m, ss[t * 128 + j]);
      float l = 0.f;
      for (int j = 0; j < 128; ++j) { float p = __expf(ss[t * 128 + j] - m); ss[t * 128 + j] = p; l += p; }
      ms[t] = m; ls[t] = l;
    } else {
      for (int j = 0; j < 128; ++j) ss[t * 128 + j] = 0.f;
    }
  }
  __syncthreads();
  // P @ V : 7 row tiles x 2 col tiles, K=128 (4 chunks)
  for (int tix = w; tix < 14; tix += 8) {
    int rt = tix >> 1, ctv = tix & 1;
    F32x8 acc;
#pragma unroll
    for (int j = 0; j < 8; ++j) acc.e[j] = 0.f;
#pragma unroll
    for (int kc = 0; kc < 4; ++kc) {
      v16bf a = afrag_ldsf(ss, rt * 16, 128, kc * 32);
      v16bf bm = bfrag_vglobal(Vb, jbase, kc * 32, hoff, ctv * 16, Np);
      acc.v = wmma_bf16(a, bm, acc.v);
    }
    int colc = ctv * 16 + (lane & 15);
    int row0 = rt * 16 + ((lane >> 4) << 3);
#pragma unroll
    for (int r = 0; r < 8; ++r) {
      int row = row0 + r;
      if (row < NQn)
        part[(((size_t)bh * NCH + cidx) * NQn + row) * 32 + colc] = acc.e[r];
    }
  }
  if (t < NQn) {
    pm[((size_t)bh * NCH + cidx) * NQn + t] = ms[t];
    pl[((size_t)bh * NCH + cidx) * NQn + t] = ls[t];
  }
}

// ---------------- Combine partials ----------------
__global__ void k_ca_combine(const float* __restrict__ part, const float* __restrict__ pm,
                             const float* __restrict__ pl, float* __restrict__ out) {
  int t = threadIdx.x;
  int bq = blockIdx.x;                 // b*NQ + q
  int b = bq / NQn, q = bq % NQn;
  int h = t >> 5, d = t & 31;
  int bh = b * Hh + h;
  size_t base = ((size_t)bh * NCH) * NQn + q;
  float m = -1e30f;
  for (int c = 0; c < NCH; ++c) m = fmaxf(m, pm[base + (size_t)c * NQn]);
  float L = 0.f, acc = 0.f;
  for (int c = 0; c < NCH; ++c) {
    size_t idx = base + (size_t)c * NQn;
    float wgt = __expf(pm[idx] - m);
    L += pl[idx] * wgt;
    acc += wgt * part[idx * 32 + d];
  }
  out[(size_t)bq * Dm + h * 32 + d] = acc / L;
}

// ---------------- Self attention (tiny, per b*h block) ----------------
__global__ void k_sa_attn(const float* __restrict__ qkv, float* __restrict__ out) {
  __shared__ float Ks[100 * 32];
  __shared__ float Vs[100 * 32];
  int t = threadIdx.x;                 // 128 threads
  int bh = blockIdx.x;
  int b = bh >> 3, h = bh & 7;
  int hoff = h * 32;
  for (int i = t; i < 100 * 32; i += 128) {
    int q = i >> 5, d = i & 31;
    size_t rb = ((size_t)(b * NQn + q)) * 768;
    Ks[i] = qkv[rb + 256 + hoff + d];
    Vs[i] = qkv[rb + 512 + hoff + d];
  }
  __syncthreads();
  if (t < NQn) {
    float qr[32];
    size_t rb = ((size_t)(b * NQn + t)) * 768;
#pragma unroll
    for (int d = 0; d < 32; ++d) qr[d] = qkv[rb + hoff + d] * SCALE;
    float m = -1e30f;
    for (int j = 0; j < 100; ++j) {
      float s = 0.f;
#pragma unroll
      for (int d = 0; d < 32; ++d) s += qr[d] * Ks[j * 32 + d];
      m = fmaxf(m, s);
    }
    float l = 0.f;
    float acc[32];
#pragma unroll
    for (int d = 0; d < 32; ++d) acc[d] = 0.f;
    for (int j = 0; j < 100; ++j) {
      float s = 0.f;
#pragma unroll
      for (int d = 0; d < 32; ++d) s += qr[d] * Ks[j * 32 + d];
      float p = __expf(s - m);
      l += p;
#pragma unroll
      for (int d = 0; d < 32; ++d) acc[d] += p * Vs[j * 32 + d];
    }
    float rl = 1.f / l;
#pragma unroll
    for (int d = 0; d < 32; ++d)
      out[((size_t)(b * NQn + t)) * Dm + hoff + d] = acc[d] * rl;
  }
}

// ---------------- pred_masks = qn @ mask_feats^T (WMMA) ----------------
__global__ void k_masks(const __bf16* __restrict__ qn, const __bf16* __restrict__ maskf,
                        float* __restrict__ out) {
  __shared__ __bf16 qs[112 * 256];
  int t = threadIdx.x;
  int nb = (Np + 255) / 256;          // 98
  int b = blockIdx.x / nb;
  int nbase = (blockIdx.x % nb) * 256;
  {
    unsigned int* dst = (unsigned int*)qs;
    const unsigned int* src = (const unsigned int*)qn;
    for (int i = t; i < 112 * 128; i += 256) {
      int q = i >> 7;
      unsigned int v = 0u;
      if (q < NQn) v = src[((size_t)(b * NQn + q)) * 128 + (i & 127)];
      dst[i] = v;
    }
  }
  __syncthreads();
  int w = t >> 5, lane = t & 31;
  const __bf16* mbase = maskf + ((size_t)b * Np + nbase) * Dm;
  for (int tix = w; tix < 112; tix += 8) {
    int rt = tix >> 4, ct = tix & 15;
    F32x8 acc;
#pragma unroll
    for (int j = 0; j < 8; ++j) acc.e[j] = 0.f;
#pragma unroll
    for (int kc = 0; kc < 8; ++kc) {
      v16bf a = afrag_lds(qs, rt * 16, 256, kc * 32);
      v16bf bm = bfrag_rowk(mbase, ct * 16, Dm, kc * 32, Np - nbase);
      acc.v = wmma_bf16(a, bm, acc.v);
    }
    int colc = ct * 16 + (lane & 15);
    int n = nbase + colc;
    int row0 = rt * 16 + ((lane >> 4) << 3);
    if (n < Np) {
#pragma unroll
      for (int r = 0; r < 8; ++r) {
        int row = row0 + r;
        if (row < NQn)
          out[((size_t)(b * NQn + row)) * Np + n] = acc.e[r];
      }
    }
  }
}

// ---------------- small utility kernels ----------------
__global__ void k_bcast_query(const float* __restrict__ qe, float* __restrict__ query) {
  int t = threadIdx.x;
  int bq = blockIdx.x;
  int q = bq % NQn;
  query[(size_t)bq * Dm + t] = qe[(size_t)q * Dm + t];
}

__global__ void k_cvt_bf16(const float* __restrict__ in, __bf16* __restrict__ outp, int n) {
  int i = blockIdx.x * 256 + threadIdx.x;
  if (i < n) outp[i] = (__bf16)in[i];
}

// =============================== host ===============================
extern "C" void kernel_launch(void* const* d_in, const int* in_sizes, int n_in,
                              void* d_out, int out_size, void* d_ws, size_t ws_size,
                              hipStream_t stream) {
  (void)in_sizes; (void)n_in; (void)out_size; (void)ws_size;
  const float* x       = (const float*)d_in[0];
  const float* ipw     = (const float*)d_in[2];
  const float* ipb     = (const float*)d_in[3];
  const float* iplng   = (const float*)d_in[4];
  const float* iplnb   = (const float*)d_in[5];
  const float* xmw1    = (const float*)d_in[6];
  const float* xmb1    = (const float*)d_in[7];
  const float* xmw2    = (const float*)d_in[8];
  const float* xmb2    = (const float*)d_in[9];
  const float* qemb    = (const float*)d_in[10];
  const float* ca_wqkv = (const float*)d_in[11];
  const float* ca_bqkv = (const float*)d_in[12];
  const float* ca_wo   = (const float*)d_in[13];
  const float* ca_bo   = (const float*)d_in[14];
  const float* sa_wqkv = (const float*)d_in[15];
  const float* sa_bqkv = (const float*)d_in[16];
  const float* sa_wo   = (const float*)d_in[17];
  const float* sa_bo   = (const float*)d_in[18];
  const float* salng   = (const float*)d_in[19];
  const float* salnb   = (const float*)d_in[20];
  const float* fw1     = (const float*)d_in[21];
  const float* fb1     = (const float*)d_in[22];
  const float* fw2     = (const float*)d_in[23];
  const float* fb2     = (const float*)d_in[24];
  const float* flng    = (const float*)d_in[25];
  const float* flnb    = (const float*)d_in[26];
  const float* olng    = (const float*)d_in[27];
  const float* olnb    = (const float*)d_in[28];
  const float* cw1     = (const float*)d_in[29];
  const float* cb1     = (const float*)d_in[30];
  const float* cw2     = (const float*)d_in[31];
  const float* cb2     = (const float*)d_in[32];

  char* ws = (char*)d_ws;
  size_t off = 0;
  auto carve = [&](size_t bytes) -> void* {
    void* p = ws + off;
    off = (off + bytes + 255) & ~(size_t)255;
    return p;
  };
  __bf16* inst   = (__bf16*)carve((size_t)BN * Dm * 2);
  __bf16* maskf  = (__bf16*)carve((size_t)BN * Dm * 2);
  __bf16* Kb     = (__bf16*)carve((size_t)BN * Dm * 2);
  __bf16* Vb     = (__bf16*)carve((size_t)BN * Dm * 2);
  float*  part   = (float*)carve((size_t)Bb * Hh * NCH * NQn * 32 * 4);
  float*  pm     = (float*)carve((size_t)Bb * Hh * NCH * NQn * 4);
  float*  pl     = (float*)carve((size_t)Bb * Hh * NCH * NQn * 4);
  float*  query  = (float*)carve((size_t)Bb * NQn * Dm * 4);
  float*  qlin   = (float*)carve((size_t)Bb * NQn * Dm * 4);
  float*  attn   = (float*)carve((size_t)Bb * NQn * Dm * 4);
  float*  qkvs   = (float*)carve((size_t)Bb * NQn * 768 * 4);
  float*  saout  = (float*)carve((size_t)Bb * NQn * Dm * 4);
  float*  hidden = (float*)carve((size_t)Bb * NQn * HID * 4);
  float*  qn     = (float*)carve((size_t)Bb * NQn * Dm * 4);
  __bf16* qnb    = (__bf16*)carve((size_t)Bb * NQn * Dm * 2);
  float*  clst   = (float*)carve((size_t)Bb * NQn * Dm * 4);

  const int NR = Bb * NQn;   // 400 query rows

  // Stage 1: input proj + mask feats
  k_inproj<<<BN / 16, 256, 0, stream>>>(x, ipw, ipb, iplng, iplnb, xmw1, xmb1, xmw2, xmb2, inst, maskf);
  k_bcast_query<<<NR, 256, 0, stream>>>(qemb, query);

  for (int i = 0; i < Ll; ++i) {
    const float* wq = ca_wqkv + (size_t)i * 768 * 256;
    const float* wk = wq + 256 * 256;
    const float* wv = wk + 256 * 256;
    const float* bq = ca_bqkv + (size_t)i * 768;
    const float* bk = bq + 256;
    const float* bv = bq + 512;
    const float* wo = ca_wo + (size_t)i * 256 * 256;
    const float* bo = ca_bo + (size_t)i * 256;

    // K/V projection of inst (big WMMA GEMM)
    k_kvproj<<<(BN + 63) / 64, 256, 0, stream>>>(inst, wk, wv, bk, bv, Kb, Vb);
    // Q projection (small)
    k_linear<<<NR, 256, 0, stream>>>(query, 256, wq, bq, nullptr, nullptr, nullptr, 256, 0, qlin);
    // flash cross-attention: partials + combine
    k_ca_partial<<<Bb * Hh * NCH, 256, 0, stream>>>(qlin, Kb, Vb, part, pm, pl);
    k_ca_combine<<<NR, 256, 0, stream>>>(part, pm, pl, attn);
    // CA out-proj + pure residual (no norm, matches reference)
    k_linear<<<NR, 256, 0, stream>>>(attn, 256, wo, bo, query, nullptr, nullptr, 256, 0, query);

    // Self-attention
    const float* swqkv = sa_wqkv + (size_t)i * 768 * 256;
    const float* sbqkv = sa_bqkv + (size_t)i * 768;
    const float* swo   = sa_wo + (size_t)i * 256 * 256;
    const float* sbo   = sa_bo + (size_t)i * 256;
    k_linear<<<NR, 256, 0, stream>>>(query, 256, swqkv, sbqkv, nullptr, nullptr, nullptr, 768, 0, qkvs);
    k_sa_attn<<<Bb * Hh, 128, 0, stream>>>(qkvs, saout);
    k_linear<<<NR, 256, 0, stream>>>(saout, 256, swo, sbo, query,
                                     salng + (size_t)i * 256, salnb + (size_t)i * 256, 256, 0, query);

    // FFN
    k_linear<<<NR, 256, 0, stream>>>(query, 256, fw1 + (size_t)i * HID * 256, fb1 + (size_t)i * HID,
                                     nullptr, nullptr, nullptr, HID, 1, hidden);
    k_linear<<<NR, 256, 0, stream>>>(hidden, HID, fw2 + (size_t)i * 256 * HID, fb2 + (size_t)i * 256,
                                     query, flng + (size_t)i * 256, flnb + (size_t)i * 256, 256, 0, query);
  }

  // Output heads
  k_linear<<<NR, 256, 0, stream>>>(query, 256, nullptr, nullptr, nullptr, olng, olnb, 256, 0, qn);
  k_cvt_bf16<<<(NR * Dm + 255) / 256, 256, 0, stream>>>(qn, qnb, NR * Dm);
  k_linear<<<NR, 256, 0, stream>>>(qn, 256, cw1, cb1, nullptr, nullptr, nullptr, 256, 1, clst);
  k_linear<<<NR, 256, 0, stream>>>(clst, 256, cw2, cb2, nullptr, nullptr, nullptr, NCo, 0, (float*)d_out);
  k_masks<<<Bb * ((Np + 255) / 256), 256, 0, stream>>>(qnb, maskf, (float*)d_out + (size_t)NR * NCo);
}
